// DynamicsPredictor_34308198760589
// MI455X (gfx1250) — matrix operand order repaired and checked by reading, not compiled, not run on hardware
//
#include <hip/hip_runtime.h>
#include <hip/hip_bf16.h>
#include <cstdint>
#include <cstddef>

typedef _Float16 h16;
typedef __attribute__((ext_vector_type(16))) _Float16 v16h_t;
typedef __attribute__((ext_vector_type(8)))  _Float16 v8h_t;
typedef __attribute__((ext_vector_type(8)))  float    v8f_t;

#define BB 2
#define NNODE 1024
#define NP 960
#define NRELC 16384
#define NI 4

// ---------------------------------------------------------------------------
// Fused GEMM: C = act(A @ W + b [+ resid]) with f16 WMMA, 4-way N-blocking.
// One wave computes a 16x64 strip of C (4 accumulators); the A fragment is
// loaded once per K-step and reused for 4 WMMAs -> 4x less A traffic.
// A: f16 row-major [M x lda], K padded to multiple of 32 (pad cols may be
// garbage; corresponding Wt rows are zero so they contribute nothing).
// Wt: f16 [Npad64 rows x kpad], Wt[n*kpad + k] = W[k][n] (pre-transposed,
// zero-padded in both n and k so all fragment loads are in-bounds).
// A fragment (16x32 f16): lane L, row m0+L%16; halves e=0..15 are
// k = 32t + 16*(L>=16) + e, one contiguous 32B chunk of A row m (same order as B).
// B fragment (32x16): lane L, col n+L%16; halves e=0..15 are
// k = 32t + 16*(L>=16) + e  -> one contiguous 32B chunk of Wt row n.
// C/D: lane L, vgpr j -> (m = m0 + j + 8*(L>=16), n = n0 + L%16).
// ---------------------------------------------------------------------------
__global__ __launch_bounds__(32) void k_gemm(
    const h16* __restrict__ A, int lda,
    const h16* __restrict__ Wt,
    const float* __restrict__ bias, int nout,
    const float* __restrict__ resid, int ldres,
    h16* __restrict__ C16, int ldc16,
    float* __restrict__ C32, int ldc32,
    int kpad, int do_relu)
{
  const int lane = threadIdx.x;
  const int l16  = lane & 15;
  const int hi   = lane >> 4;            // 0 or 1
  const int m0   = blockIdx.x * 16;
  const int n0   = blockIdx.y * 64;

  const h16* __restrict__ arow = A  + (size_t)(m0 + l16) * lda;
  const h16* __restrict__ brow = Wt + (size_t)(n0 + l16) * kpad;
  const size_t bstride = (size_t)16 * kpad;   // between 16-col tiles

  v8f_t acc0 = {}, acc1 = {}, acc2 = {}, acc3 = {};
  for (int k = 0; k < kpad; k += 32) {
    const int ako = k + (hi << 4);
    v8h_t a0 = *(const v8h_t*)(arow + ako);
    v8h_t a1 = *(const v8h_t*)(arow + ako + 8);
    v16h_t av = __builtin_shufflevector(a0, a1, 0,1,2,3,4,5,6,7,8,9,10,11,12,13,14,15);
    const int bko = k + (hi << 4);

    v8h_t b0, b1;
    v16h_t bv;
    b0 = *(const v8h_t*)(brow + 0*bstride + bko);
    b1 = *(const v8h_t*)(brow + 0*bstride + bko + 8);
    bv = __builtin_shufflevector(b0, b1, 0,1,2,3,4,5,6,7,8,9,10,11,12,13,14,15);
    acc0 = __builtin_amdgcn_wmma_f32_16x16x32_f16(false, av, false, bv, (short)0, acc0, false, false);

    b0 = *(const v8h_t*)(brow + 1*bstride + bko);
    b1 = *(const v8h_t*)(brow + 1*bstride + bko + 8);
    bv = __builtin_shufflevector(b0, b1, 0,1,2,3,4,5,6,7,8,9,10,11,12,13,14,15);
    acc1 = __builtin_amdgcn_wmma_f32_16x16x32_f16(false, av, false, bv, (short)0, acc1, false, false);

    b0 = *(const v8h_t*)(brow + 2*bstride + bko);
    b1 = *(const v8h_t*)(brow + 2*bstride + bko + 8);
    bv = __builtin_shufflevector(b0, b1, 0,1,2,3,4,5,6,7,8,9,10,11,12,13,14,15);
    acc2 = __builtin_amdgcn_wmma_f32_16x16x32_f16(false, av, false, bv, (short)0, acc2, false, false);

    b0 = *(const v8h_t*)(brow + 3*bstride + bko);
    b1 = *(const v8h_t*)(brow + 3*bstride + bko + 8);
    bv = __builtin_shufflevector(b0, b1, 0,1,2,3,4,5,6,7,8,9,10,11,12,13,14,15);
    acc3 = __builtin_amdgcn_wmma_f32_16x16x32_f16(false, av, false, bv, (short)0, acc3, false, false);
  }

  v8f_t accs[4] = {acc0, acc1, acc2, acc3};
  #pragma unroll
  for (int t = 0; t < 4; ++t) {
    const int n = n0 + t * 16 + l16;
    if (n < nout) {
      const float bz = bias ? bias[n] : 0.0f;
      #pragma unroll
      for (int j = 0; j < 8; ++j) {
        const int m = m0 + j + (hi << 3);
        float v = accs[t][j] + bz;
        if (resid) v += resid[(size_t)m * ldres + n];
        if (do_relu) v = v > 0.0f ? v : 0.0f;
        if (C16) C16[(size_t)m * ldc16 + n] = (h16)v;
        if (C32) C32[(size_t)m * ldc32 + n] = v;
      }
    }
  }
}

// Weight transpose + f32->f16 + zero pad. Wt[n*kpad+k] = W[k*nout+n].
__global__ void k_wt(const float* __restrict__ W, h16* __restrict__ Wt,
                     int K, int nout, int kpad, int npad)
{
  int t = blockIdx.x * blockDim.x + threadIdx.x;
  if (t >= npad * kpad) return;
  int n = t / kpad, k = t % kpad;
  float v = (n < nout && k < K) ? W[(size_t)k * nout + n] : 0.0f;
  Wt[t] = (h16)v;
}

// One wave per one-hot row: recover index; store global node row b*NNODE+col.
__global__ __launch_bounds__(32) void k_onehot_idx(const float* __restrict__ R,
                                                   int* __restrict__ idx,
                                                   int rows_per_b)
{
  int row = blockIdx.x;
  int lane = threadIdx.x;
  const float* r = R + (size_t)row * NNODE;
  int local = -1;
  for (int c = lane; c < NNODE; c += 32)
    if (r[c] > 0.5f) local = c;
  for (int o = 16; o > 0; o >>= 1) {
    int other = __shfl_xor(local, o, 32);
    local = local > other ? local : other;
  }
  if (lane == 0) idx[row] = (row / rows_per_b) * NNODE + (local < 0 ? 0 : local);
}

__global__ void k_inst(const float* __restrict__ pi, const float* __restrict__ p_rigid,
                       int* __restrict__ inst_idx, float* __restrict__ prpp)
{
  int t = blockIdx.x * blockDim.x + threadIdx.x;
  if (t >= BB * NP) return;
  int b = t / NP;
  const float* row = pi + (size_t)t * NI;
  int bi = 0;
  for (int i = 0; i < NI; ++i) if (row[i] > 0.5f) bi = i;
  inst_idx[t] = bi;
  prpp[t] = p_rigid[b * NI + bi];
}

// snt[b,n,12]: d<9 -> ((state[h+1]-state[h])-mean_d)/std_d ; d>=9 -> (cur-mean_p)/std_p
__global__ void k_snt(const float* __restrict__ state,
                      const float* __restrict__ mean_p, const float* __restrict__ std_p,
                      const float* __restrict__ mean_d, const float* __restrict__ std_d,
                      float* __restrict__ snt)
{
  int t = blockIdx.x * blockDim.x + threadIdx.x;
  if (t >= BB * NNODE * 12) return;
  int d = t % 12, n = (t / 12) % NNODE, b = t / (12 * NNODE);
  float v;
  if (d < 9) {
    int h = d / 3, s = d % 3;
    float a = state[(((size_t)b*4 + h+1)*NNODE + n)*3 + s];
    float c = state[(((size_t)b*4 + h  )*NNODE + n)*3 + s];
    v = (a - c - mean_d[s]) / std_d[s];
  } else {
    int s = d - 9;
    v = (state[(((size_t)b*4 + 3)*NNODE + n)*3 + s] - mean_p[s]) / std_p[s];
  }
  snt[t] = v;
}

__global__ void k_zero_f32(float* __restrict__ p, int n)
{
  int t = blockIdx.x * blockDim.x + threadIdx.x;
  if (t < n) p[t] = 0.0f;
}

__global__ void k_center_acc(const float* __restrict__ snt, const int* __restrict__ inst_idx,
                             float* __restrict__ center, float* __restrict__ ccnt)
{
  int t = blockIdx.x;                 // b*NP+p
  int d = threadIdx.x; if (d >= 12) return;
  int b = t / NP, n = t % NP;
  int i = inst_idx[t];
  atomicAdd(&center[(b*NI + i)*12 + d], snt[((size_t)b*NNODE + n)*12 + d]);
  if (d == 0) atomicAdd(&ccnt[b*NI + i], 1.0f);
}

__global__ void k_center_norm(float* __restrict__ center, const float* __restrict__ ccnt)
{
  int t = blockIdx.x * blockDim.x + threadIdx.x;
  if (t >= BB * NI * 12) return;
  center[t] /= (ccnt[t / 12] + 1e-6f);
}

// node_feat row (448 wide, f16): [attrs(3), phys(1), offset(12), memory(400), snt(12), pad(20)]
__global__ void k_node_feat(const float* __restrict__ attrs, const float* __restrict__ phys,
                            const float* __restrict__ mem, const float* __restrict__ snt,
                            const float* __restrict__ prpp, const int* __restrict__ inst_idx,
                            const float* __restrict__ center, h16* __restrict__ nf)
{
  int row = blockIdx.x;               // b*NNODE+n
  int b = row / NNODE, n = row % NNODE;
  h16* dst = nf + (size_t)row * 448;
  for (int c = threadIdx.x; c < 448; c += blockDim.x) {
    float v = 0.0f;
    if (c < 3) v = attrs[(size_t)row*3 + c];
    else if (c == 3) v = (n < NP) ? phys[b*NP + n] : 0.0f;
    else if (c < 16) {
      int d = c - 4;
      if (n < NP) {
        float pr = prpp[b*NP + n];
        int i = inst_idx[b*NP + n];
        v = pr * (snt[(size_t)row*12 + d] - center[(b*NI + i)*12 + d]);
      }
    } else if (c < 416) {
      int m = c - 16;
      v = mem[((size_t)b*400 + m)*NNODE + n];
    } else if (c < 428) {
      v = snt[(size_t)row*12 + (c - 416)];
    }
    dst[c] = (h16)v;
  }
}

// relation input row (864 wide): [attrs_r(416), attrs_s(416), snt_r(12), snt_s(12), gdiff(1), pad(7)]
__global__ void k_build_rel(const h16* __restrict__ nf, const int* __restrict__ idx_r,
                            const int* __restrict__ idx_s, const int* __restrict__ inst_idx,
                            h16* __restrict__ rel)
{
  int r = blockIdx.x;
  int rr = idx_r[r], ss = idx_s[r];
  const h16* fr = nf + (size_t)rr * 448;
  const h16* fs = nf + (size_t)ss * 448;
  int b = rr / NNODE, nr_ = rr % NNODE, ns_ = ss % NNODE;
  bool pr = nr_ < NP, ps = ns_ < NP;
  float gd;
  if (pr && ps) gd = (inst_idx[b*NP + nr_] == inst_idx[b*NP + ns_]) ? 0.0f : 2.0f;
  else gd = (pr != ps) ? 1.0f : 0.0f;
  h16* dst = rel + (size_t)r * 864;
  for (int c = threadIdx.x; c < 864; c += blockDim.x) {
    float v;
    if (c < 416) v = (float)fr[c];
    else if (c < 832) v = (float)fs[c - 416];
    else if (c < 844) v = (float)fr[416 + c - 832];
    else if (c < 856) v = (float)fs[416 + c - 844];
    else if (c == 856) v = gd;
    else v = 0.0f;
    dst[c] = (h16)v;
  }
}

// effcat row (608 wide): [relation_encode(200), pe[recv](200), pe[send](200), pad(8)]
__global__ void k_effcat(const h16* __restrict__ renc, const h16* __restrict__ pe,
                         const int* __restrict__ idx_r, const int* __restrict__ idx_s,
                         h16* __restrict__ cat)
{
  int r = blockIdx.x;
  int rr = idx_r[r], ss = idx_s[r];
  h16* dst = cat + (size_t)r * 608;
  for (int c = threadIdx.x; c < 608; c += blockDim.x) {
    float v = 0.0f;
    if (c < 200) v = (float)renc[(size_t)r*224 + c];
    else if (c < 400) v = (float)pe[(size_t)rr*224 + (c - 200)];
    else if (c < 600) v = (float)pe[(size_t)ss*224 + (c - 400)];
    dst[c] = (h16)v;
  }
}

// agg[recv] += eff_rel[r]   (Rr^T @ eff_rel, Rr one-hot)
__global__ void k_agg(const h16* __restrict__ effrel, const int* __restrict__ idx_r,
                      float* __restrict__ agg)
{
  int r = blockIdx.x, c = threadIdx.x;
  if (c >= 200) return;
  int rr = idx_r[r];
  atomicAdd(&agg[(size_t)rr*200 + c], (float)effrel[(size_t)r*224 + c]);
}

// pp input row (416 wide): [particle_encode(200), agg(200), pad(16)]
__global__ void k_pp_in(const h16* __restrict__ penc, const float* __restrict__ agg,
                        h16* __restrict__ dst)
{
  int row = blockIdx.x;
  for (int c = threadIdx.x; c < 416; c += blockDim.x) {
    float v = 0.0f;
    if (c < 200) v = (float)penc[(size_t)row*224 + c];
    else if (c < 400) v = agg[(size_t)row*200 + (c - 200)];
    dst[(size_t)row*416 + c] = (h16)v;
  }
}

__global__ void k_copy_h(const h16* __restrict__ s, h16* __restrict__ d, int n)
{
  int t = blockIdx.x * blockDim.x + threadIdx.x;
  if (t < n) d[t] = s[t];
}

// inst_eff[b,i] += pe[b, p]   (p_instance^T @ pe)
__global__ void k_inst_eff(const h16* __restrict__ pe, const int* __restrict__ inst_idx,
                           float* __restrict__ ie)
{
  int t = blockIdx.x, c = threadIdx.x;
  if (c >= 200) return;
  int b = t / NP, p = t % NP;
  int i = inst_idx[t];
  atomicAdd(&ie[(b*NI + i)*200 + c], (float)pe[((size_t)b*NNODE + p)*224 + c]);
}

__global__ void k_inst_stage(const float* __restrict__ ie, h16* __restrict__ dst)
{
  int t = blockIdx.x * blockDim.x + threadIdx.x;
  if (t >= 16 * 224) return;
  int r = t / 224, c = t % 224;
  float v = (r < BB*NI && c < 200) ? ie[r*200 + c] : 0.0f;
  dst[t] = (h16)v;
}

// quat -> rotation matrix, translation, center (8 instances total)
__global__ void k_rigidmats(const float* __restrict__ irp, const float* __restrict__ center,
                            const float* __restrict__ std_p, const float* __restrict__ mean_p,
                            const float* __restrict__ std_d, const float* __restrict__ mean_d,
                            float* __restrict__ Rm, float* __restrict__ T, float* __restrict__ C2)
{
  int t = threadIdx.x;
  if (t >= BB * NI) return;
  const float* q0 = irp + t * 8;
  float w = q0[0] + 1.0f, x = q0[1], y = q0[2], z = q0[3];
  float inv = 1.0f / sqrtf(w*w + x*x + y*y + z*z);
  w *= inv; x *= inv; y *= inv; z *= inv;
  float* R = Rm + t * 9;
  R[0] = 1.f - 2.f*(y*y + z*z); R[1] = 2.f*(x*y + z*w);       R[2] = 2.f*(x*z - y*w);
  R[3] = 2.f*(x*y - z*w);       R[4] = 1.f - 2.f*(x*x + z*z); R[5] = 2.f*(y*z + x*w);
  R[6] = 2.f*(x*z + y*w);       R[7] = 2.f*(y*z - x*w);       R[8] = 1.f - 2.f*(x*x + y*y);
  for (int s = 0; s < 3; ++s) {
    T[t*3 + s]  = q0[4 + s] * std_d[s] + mean_d[s];
    C2[t*3 + s] = center[t*12 + 9 + s] * std_p[s] + mean_p[s];
  }
}

__global__ void k_finalize(const float* __restrict__ state, const float* __restrict__ nr_out,
                           const float* __restrict__ prpp, const int* __restrict__ inst_idx,
                           const float* __restrict__ p_rigid,
                           const float* __restrict__ Rm, const float* __restrict__ T,
                           const float* __restrict__ C2,
                           const float* __restrict__ std_d, const float* __restrict__ mean_d,
                           float* __restrict__ out)
{
  int t = blockIdx.x * blockDim.x + threadIdx.x;
  if (t >= BB * NP) return;
  int b = t / NP, p = t % NP;
  int i = inst_idx[t];
  int ti = b*NI + i;
  float p0[3], pc[3];
  for (int s = 0; s < 3; ++s) p0[s] = state[(((size_t)b*4 + 3)*NNODE + p)*3 + s];
  for (int s = 0; s < 3; ++s) pc[s] = p0[s] - C2[ti*3 + s];
  float pr = prpp[t];
  for (int d = 0; d < 3; ++d) {
    float p1 = pc[0]*Rm[ti*9 + 0 + d] + pc[1]*Rm[ti*9 + 3 + d] + pc[2]*Rm[ti*9 + 6 + d]
             + T[ti*3 + d] + C2[ti*3 + d];
    float rm = ((p1 - p0[d]) - mean_d[d]) / std_d[d];
    float nr = nr_out[((size_t)b*NNODE + p)*4 + d];
    float motion = (1.0f - pr) * nr + p_rigid[b*NI + i] * rm;
    out[(size_t)BB*NP*3 + (size_t)t*3 + d] = motion;                // pred_motion
    out[(size_t)t*3 + d] = p0[d] + motion * std_d[d] + mean_d[d];   // pred_pos
  }
}

extern "C" void kernel_launch(void* const* d_in, const int* in_sizes, int n_in,
                              void* d_out, int out_size, void* d_ws, size_t ws_size,
                              hipStream_t stream)
{
  const float* attrs   = (const float*)d_in[0];
  const float* state   = (const float*)d_in[1];
  const float* Rr      = (const float*)d_in[2];
  const float* Rs      = (const float*)d_in[3];
  const float* memory  = (const float*)d_in[4];
  const float* p_rigid = (const float*)d_in[5];
  const float* p_inst  = (const float*)d_in[6];
  const float* physp   = (const float*)d_in[7];
  const float* mean_p  = (const float*)d_in[8];
  const float* std_p   = (const float*)d_in[9];
  const float* mean_d  = (const float*)d_in[10];
  const float* std_d   = (const float*)d_in[11];

  // params leaf order: detect insertion order (pe first, W0 428x200=85600)
  // vs jax sorted-key order (nonrigid first, W0 200x200=40000).
  int ipe, ire, ipp, irp0, irg, inr;
  if (n_in > 12 && in_sizes[12] == 428 * 200) { ipe = 12; ire = 18; ipp = 24; irp0 = 26; irg = 28; inr = 34; }
  else                                        { inr = 12; ipe = 18; ipp = 24; ire = 26; irg = 32; irp0 = 38; }
  auto F = [&](int i) { return (const float*)d_in[i]; };

  // ----- workspace layout -----
  size_t off = 0;
  auto alloc = [&](size_t bytes) -> void* {
    off = (off + 255) & ~(size_t)255;
    void* p = (char*)d_ws + off;
    off += bytes;
    return p;
  };
  int*   idx_r    = (int*)  alloc((size_t)BB*NRELC*4);
  int*   idx_s    = (int*)  alloc((size_t)BB*NRELC*4);
  int*   inst_idx = (int*)  alloc((size_t)BB*NP*4);
  float* prpp     = (float*)alloc((size_t)BB*NP*4);
  float* snt      = (float*)alloc((size_t)BB*NNODE*12*4);
  float* center   = (float*)alloc((size_t)BB*NI*12*4);
  float* ccnt     = (float*)alloc((size_t)BB*NI*4);
  h16*   nodef    = (h16*)  alloc((size_t)BB*NNODE*448*2);
  h16*   rel_in   = (h16*)  alloc((size_t)BB*NRELC*864*2);  // reused as effcat (608)
  h16*   reH1     = (h16*)  alloc((size_t)BB*NRELC*320*2);  // reused as eff_rel (224)
  h16*   reH2     = (h16*)  alloc((size_t)BB*NRELC*320*2);
  h16*   rel_enc  = (h16*)  alloc((size_t)BB*NRELC*224*2);
  h16*   pe_h1    = (h16*)  alloc((size_t)BB*NNODE*224*2);  // reused as nonrigid h1
  h16*   pe_h2    = (h16*)  alloc((size_t)BB*NNODE*224*2);  // reused as nonrigid h2
  h16*   pcl_enc  = (h16*)  alloc((size_t)BB*NNODE*224*2);
  h16*   pe16     = (h16*)  alloc((size_t)BB*NNODE*224*2);
  float* pe32     = (float*)alloc((size_t)BB*NNODE*200*4);
  float* agg      = (float*)alloc((size_t)BB*NNODE*200*4);
  h16*   pp_in    = (h16*)  alloc((size_t)BB*NNODE*416*2);
  float* nr_out   = (float*)alloc((size_t)BB*NNODE*4*4);
  float* inst_eff = (float*)alloc((size_t)BB*NI*200*4);
  h16*   inst16   = (h16*)  alloc((size_t)16*224*2);
  h16*   ri_h1    = (h16*)  alloc((size_t)16*224*2);
  h16*   ri_h2    = (h16*)  alloc((size_t)16*224*2);
  float* irp      = (float*)alloc((size_t)16*8*4);
  float* Rmat     = (float*)alloc((size_t)BB*NI*9*4);
  float* Tvec     = (float*)alloc((size_t)BB*NI*3*4);
  float* C2v      = (float*)alloc((size_t)BB*NI*3*4);
  h16* effcat  = rel_in;   // alias (rel_in dead after relation layer 1)
  h16* eff_rel = reH1;     // alias (reH1 dead after relation layer 2)

  // staged transposed weights (N padded to multiple of 64 for 16x64 strips)
  auto wtbuf = [&](int kpad, int npad) { return (h16*)alloc((size_t)kpad * npad * 2); };
  h16 *peW0 = wtbuf(448,256), *peW1 = wtbuf(224,256), *peW2 = wtbuf(224,256);
  h16 *reW0 = wtbuf(864,320), *reW1 = wtbuf(320,320), *reW2 = wtbuf(320,256);
  h16 *ppW  = wtbuf(416,256), *rpW  = wtbuf(608,256);
  h16 *rgW0 = wtbuf(224,256), *rgW1 = wtbuf(224,256), *rgW2 = wtbuf(224,64);
  h16 *nrW0 = wtbuf(224,256), *nrW1 = wtbuf(224,256), *nrW2 = wtbuf(224,64);

  auto wt = [&](const float* W, h16* dst, int K, int Nout, int Kpad) {
    int Npad = ((Nout + 63) / 64) * 64;
    int n = Npad * Kpad;
    k_wt<<<dim3((n + 255) / 256), dim3(256), 0, stream>>>(W, dst, K, Nout, Kpad, Npad);
  };
  wt(F(ipe+0), peW0, 428,200,448); wt(F(ipe+2), peW1, 200,200,224); wt(F(ipe+4), peW2, 200,200,224);
  wt(F(ire+0), reW0, 857,300,864); wt(F(ire+2), reW1, 300,300,320); wt(F(ire+4), reW2, 300,200,320);
  wt(F(ipp+0), ppW,  400,200,416); wt(F(irp0+0), rpW, 600,200,608);
  wt(F(irg+0), rgW0, 200,200,224); wt(F(irg+2), rgW1, 200,200,224); wt(F(irg+4), rgW2, 200,7,224);
  wt(F(inr+0), nrW0, 200,200,224); wt(F(inr+2), nrW1, 200,200,224); wt(F(inr+4), nrW2, 200,3,224);

  auto gemm = [&](const h16* A, int lda, const h16* Wt, int kpad,
                  const float* bias, int nout, const float* resid, int ldres,
                  h16* C16, int ldc16, float* C32, int ldc32, int M, int relu) {
    dim3 g(M / 16, (nout + 63) / 64);
    k_gemm<<<g, dim3(32), 0, stream>>>(A, lda, Wt, bias, nout, resid, ldres,
                                       C16, ldc16, C32, ldc32, kpad, relu);
  };
  auto zero = [&](float* p, int n) {
    k_zero_f32<<<dim3((n + 255) / 256), dim3(256), 0, stream>>>(p, n);
  };

  // ---- graph structure recovery ----
  k_onehot_idx<<<dim3(BB*NRELC), dim3(32), 0, stream>>>(Rr, idx_r, NRELC);
  k_onehot_idx<<<dim3(BB*NRELC), dim3(32), 0, stream>>>(Rs, idx_s, NRELC);
  k_inst<<<dim3((BB*NP + 255)/256), dim3(256), 0, stream>>>(p_inst, p_rigid, inst_idx, prpp);

  // ---- node features ----
  k_snt<<<dim3((BB*NNODE*12 + 255)/256), dim3(256), 0, stream>>>(state, mean_p, std_p, mean_d, std_d, snt);
  zero(center, BB*NI*12);
  zero(ccnt, BB*NI);
  k_center_acc<<<dim3(BB*NP), dim3(16), 0, stream>>>(snt, inst_idx, center, ccnt);
  k_center_norm<<<dim3(1), dim3(BB*NI*12), 0, stream>>>(center, ccnt);
  k_node_feat<<<dim3(BB*NNODE), dim3(128), 0, stream>>>(attrs, physp, memory, snt, prpp,
                                                        inst_idx, center, nodef);
  k_build_rel<<<dim3(BB*NRELC), dim3(128), 0, stream>>>(nodef, idx_r, idx_s, inst_idx, rel_in);

  // ---- relation encoder: 857 -> 300 -> 300 -> 200 (all ReLU) ----
  gemm(rel_in, 864, reW0, 864, F(ire+1), 300, nullptr, 0, reH1, 320, nullptr, 0, BB*NRELC, 1);
  gemm(reH1,   320, reW1, 320, F(ire+3), 300, nullptr, 0, reH2, 320, nullptr, 0, BB*NRELC, 1);
  gemm(reH2,   320, reW2, 320, F(ire+5), 200, nullptr, 0, rel_enc, 224, nullptr, 0, BB*NRELC, 1);

  // ---- particle encoder: 428 -> 200 -> 200 -> 200 (all ReLU) ----
  gemm(nodef, 448, peW0, 448, F(ipe+1), 200, nullptr, 0, pe_h1, 224, nullptr, 0, BB*NNODE, 1);
  gemm(pe_h1, 224, peW1, 224, F(ipe+3), 200, nullptr, 0, pe_h2, 224, nullptr, 0, BB*NNODE, 1);
  gemm(pe_h2, 224, peW2, 224, F(ipe+5), 200, nullptr, 0, pcl_enc, 224, pe32, 200, BB*NNODE, 1);
  k_copy_h<<<dim3((BB*NNODE*224 + 255)/256), dim3(256), 0, stream>>>(pcl_enc, pe16, BB*NNODE*224);

  // ---- propagation: 3 steps ----
  for (int step = 0; step < 3; ++step) {
    k_effcat<<<dim3(BB*NRELC), dim3(256), 0, stream>>>(rel_enc, pe16, idx_r, idx_s, effcat);
    gemm(effcat, 608, rpW, 608, F(irp0+1), 200, nullptr, 0, eff_rel, 224, nullptr, 0, BB*NRELC, 1);
    zero(agg, BB*NNODE*200);
    k_agg<<<dim3(BB*NRELC), dim3(256), 0, stream>>>(eff_rel, idx_r, agg);
    k_pp_in<<<dim3(BB*NNODE), dim3(256), 0, stream>>>(pcl_enc, agg, pp_in);
    gemm(pp_in, 416, ppW, 416, F(ipp+1), 200, pe32, 200, pe16, 224, pe32, 200, BB*NNODE, 1);
  }

  // ---- non-rigid predictor: 200 -> 200 -> 200 -> 3 (last linear) ----
  gemm(pe16,  224, nrW0, 224, F(inr+1), 200, nullptr, 0, pe_h1, 224, nullptr, 0, BB*NNODE, 1);
  gemm(pe_h1, 224, nrW1, 224, F(inr+3), 200, nullptr, 0, pe_h2, 224, nullptr, 0, BB*NNODE, 1);
  gemm(pe_h2, 224, nrW2, 224, F(inr+5),   3, nullptr, 0, nullptr, 0, nr_out, 4, BB*NNODE, 0);

  // ---- rigid predictor on instance effects ----
  zero(inst_eff, BB*NI*200);
  k_inst_eff<<<dim3(BB*NP), dim3(256), 0, stream>>>(pe16, inst_idx, inst_eff);
  k_inst_stage<<<dim3((16*224 + 255)/256), dim3(256), 0, stream>>>(inst_eff, inst16);
  gemm(inst16, 224, rgW0, 224, F(irg+1), 200, nullptr, 0, ri_h1, 224, nullptr, 0, 16, 1);
  gemm(ri_h1,  224, rgW1, 224, F(irg+3), 200, nullptr, 0, ri_h2, 224, nullptr, 0, 16, 1);
  gemm(ri_h2,  224, rgW2, 224, F(irg+5),   7, nullptr, 0, nullptr, 0, irp, 8, 16, 0);

  // ---- rigid transform + output ----
  k_rigidmats<<<dim3(1), dim3(32), 0, stream>>>(irp, center, std_p, mean_p, std_d, mean_d,
                                                Rmat, Tvec, C2v);
  k_finalize<<<dim3((BB*NP + 255)/256), dim3(256), 0, stream>>>(state, nr_out, prpp, inst_idx,
                                                                p_rigid, Rmat, Tvec, C2v,
                                                                std_d, mean_d, (float*)d_out);
  (void)ws_size; (void)out_size; (void)n_in;
}
